// CausalGraphAttention_58050777972872
// MI455X (gfx1250) — compile-verified, matrix operand order checked by reading, not run on hardware
//
#include <hip/hip_runtime.h>
#include <hip/hip_fp16.h>

typedef __attribute__((ext_vector_type(16))) _Float16 v16h;
typedef __attribute__((ext_vector_type(8)))  _Float16 v8h;
typedef __attribute__((ext_vector_type(8)))  float    v8f;
typedef __attribute__((ext_vector_type(4)))  float    v4f;

#define B_  2
#define SQ_ 2048
#define SK_ 2048
#define H_  1024
#define NH_ 16
#define HD_ 64

static __device__ inline v8f wmma16x16x32(v16h a, v16h b, v8f c) {
  // D = A(16x32 f16) * B(32x16 f16) + C(16x16 f32)
  return __builtin_amdgcn_wmma_f32_16x16x32_f16(false, a, false, b, (short)0, c,
                                                false, false);
}

// Load 16x32 f16 A/B tile in WMMA per-lane layout from row-major f16 memory
// (global or LDS). base -> element (row0, k0); ld = row stride (elements).
// ISA 7.12.2: lane (hf=lane>>4, r=lane&15) holds row r, K-chunks [8hf,+8)
// and [16+8hf,+8).
static __device__ inline v16h load_tile_f16(const _Float16* base, int ld, int lane) {
  const int hf = lane >> 4, r = lane & 15;
  const _Float16* p = base + (size_t)r * ld + 8 * hf;
  v8h c0 = *(const v8h*)(p);
  v8h c1 = *(const v8h*)(p + 16);
  v16h o;
#pragma unroll
  for (int i = 0; i < 8; ++i) { o[i] = c0[i]; o[8 + i] = c1[i]; }
  return o;
}

// CDNA5 async copy: 16B per lane, global -> LDS, tracked by ASYNCcnt.
static __device__ inline void async_g2l_b128(uint32_t lds_byte, const _Float16* gsrc) {
  asm volatile("global_load_async_to_lds_b128 %0, %1, off"
               :: "v"(lds_byte), "v"(gsrc)
               : "memory");
}

// ---------------- Stage 0: f32 -> f16 conversion ---------------------------
__global__ __launch_bounds__(256) void cvt_kernel(const float* __restrict__ src,
                                                  _Float16* __restrict__ dst,
                                                  int n8) {
  const int i = blockIdx.x * 256 + threadIdx.x;  // 8 elements per thread
  if (i < n8) {
    const v4f a = *(const v4f*)(src + (size_t)i * 8);
    const v4f b = *(const v4f*)(src + (size_t)i * 8 + 4);
    v8h o;
#pragma unroll
    for (int j = 0; j < 4; ++j) { o[j] = (_Float16)a[j]; o[4 + j] = (_Float16)b[j]; }
    *(v8h*)(dst + (size_t)i * 8) = o;
  }
}

// ---------------- Unified WMMA GEMM: Y = A @ W^T + bias --------------------
// A:[4096,1024] f16, W:[1024,1024] f16 (row n = output column n).
// mode 0: outH f16 [B,NH,S,64] (Q/K head-split)
// mode 1: outH f16 [B,NH,64,SK] (V transposed)
// mode 2: outF f32 [4096,1024]  (output projection)
// Workgroup = 4 waves, each computing 32x64; the shared 64x32 weight tile is
// staged once per workgroup via double-buffered async copies to LDS.
__global__ __launch_bounds__(128) void gemm_kernel(
    const _Float16* __restrict__ A, const _Float16* __restrict__ W,
    const float* __restrict__ bias,
    _Float16* __restrict__ outH, float* __restrict__ outF, int mode) {
  __shared__ _Float16 wtile[2][64 * 32];
  const int tid = threadIdx.x;
  const int lane = tid & 31;
  const int wave = tid >> 5;
  const int n0 = blockIdx.x * 64;
  const int m0 = blockIdx.y * 128 + wave * 32;
  // async-copy mapping: chunk c (0..255) -> 16B; row = c>>2, kchunk = c&3
  const int row0 = tid >> 2;
  const int ck8 = (tid & 3) * 8;
  const uint32_t lb0 = (uint32_t)(uintptr_t)(&wtile[0][0]);
  const uint32_t lb1 = (uint32_t)(uintptr_t)(&wtile[1][0]);

  // prologue: issue k-step 0 into buffer 0
  async_g2l_b128(lb0 + (uint32_t)tid * 16u, W + (size_t)(n0 + row0) * H_ + ck8);
  async_g2l_b128(lb0 + (uint32_t)(tid + 128) * 16u,
                 W + (size_t)(n0 + 32 + row0) * H_ + ck8);

  v8f acc[2][4] = {};
  for (int kk = 0; kk < H_; kk += 32) {
    const int cur = (kk >> 5) & 1;
    if (kk + 32 < H_) {
      const uint32_t lb = cur ? lb0 : lb1;
      async_g2l_b128(lb + (uint32_t)tid * 16u,
                     W + (size_t)(n0 + row0) * H_ + kk + 32 + ck8);
      async_g2l_b128(lb + (uint32_t)(tid + 128) * 16u,
                     W + (size_t)(n0 + 32 + row0) * H_ + kk + 32 + ck8);
      // prefetch next A tile (lowers to global_prefetch_b8)
      __builtin_prefetch(A + (size_t)(m0 + lane) * H_ + kk + 32, 0, 3);
      asm volatile("s_wait_asynccnt 0x2" ::: "memory");  // current buffer done
    } else {
      asm volatile("s_wait_asynccnt 0x0" ::: "memory");
    }
    __syncthreads();  // every wave's copies for `cur` are visible
    v16h a0 = load_tile_f16(A + (size_t)m0 * H_ + kk, H_, lane);
    v16h a1 = load_tile_f16(A + (size_t)(m0 + 16) * H_ + kk, H_, lane);
#pragma unroll
    for (int j = 0; j < 4; ++j) {
      v16h bt = load_tile_f16(&wtile[cur][16 * j * 32], 32, lane);
      acc[0][j] = wmma16x16x32(a0, bt, acc[0][j]);
      acc[1][j] = wmma16x16x32(a1, bt, acc[1][j]);
    }
    __syncthreads();  // compute done before `cur` is overwritten next round
  }

  const int hf = lane >> 4, nl = lane & 15;
#pragma unroll
  for (int t = 0; t < 2; ++t)
#pragma unroll
    for (int j = 0; j < 4; ++j) {
      const int n = n0 + 16 * j + nl;
      const int head = n >> 6, d = n & 63;
      const float bias_n = bias[n];
#pragma unroll
      for (int r = 0; r < 8; ++r) {
        const int m = m0 + 16 * t + r + 8 * hf;  // C/D layout: M = r + 8*half
        const int bb = m >> 11;                  // batch
        const int s = m & 2047;                  // sequence position
        const float val = acc[t][j][r] + bias_n;
        if (mode == 0)
          outH[(((size_t)(bb * NH_ + head) * SQ_ + s) << 6) + d] = (_Float16)val;
        else if (mode == 1)
          outH[((size_t)(bb * NH_ + head) * HD_ + d) * SK_ + s] = (_Float16)val;
        else
          outF[(size_t)m * H_ + n] = val;
      }
    }
}

// ---------------- Stage 2: causal flash attention --------------------------
__global__ __launch_bounds__(128) void attn_kernel(
    const _Float16* __restrict__ Qf,   // [B,NH,SQ,64]
    const _Float16* __restrict__ Kf,   // [B,NH,SK,64]
    const _Float16* __restrict__ Vt,   // [B,NH,64,SK]
    _Float16* __restrict__ Of) {       // [B,SQ,H] f16
  __shared__ _Float16 pbuf[4][32 * 32];  // per-wave P staging (acc->A relayout)
  const int lane = threadIdx.x & 31;
  const int wave = threadIdx.x >> 5;
  const int hh = blockIdx.y, bb = blockIdx.z;
  const int q0 = blockIdx.x * 128 + wave * 32;
  const size_t baseQ = (size_t)(bb * NH_ + hh) * SQ_ * HD_;
  const size_t baseK = (size_t)(bb * NH_ + hh) * SK_ * HD_;
  const size_t baseV = (size_t)(bb * NH_ + hh) * HD_ * SK_;

  v16h qa[2][2];
#pragma unroll
  for (int t = 0; t < 2; ++t)
#pragma unroll
    for (int kh = 0; kh < 2; ++kh)
      qa[t][kh] =
          load_tile_f16(Qf + baseQ + (size_t)(q0 + 16 * t) * HD_ + 32 * kh, HD_, lane);

  v8f o[2][4] = {};
  float rowm[2][8], rowl[2][8];
#pragma unroll
  for (int t = 0; t < 2; ++t)
#pragma unroll
    for (int r = 0; r < 8; ++r) { rowm[t][r] = -1e30f; rowl[t][r] = 0.0f; }

  const int hf = lane >> 4, nl = lane & 15;
  const float scale = 0.125f;       // 1/sqrt(64)
  const int kchunks = q0 / 32 + 1;  // causal: only kb <= q0+31 needed

  for (int c = 0; c < kchunks; ++c) {
    const int kb = c * 32;
    v8f sc[2][2];
#pragma unroll
    for (int u = 0; u < 2; ++u) {
      v16h k0 = load_tile_f16(Kf + baseK + (size_t)(kb + 16 * u) * HD_, HD_, lane);
      v16h k1 = load_tile_f16(Kf + baseK + (size_t)(kb + 16 * u) * HD_ + 32, HD_, lane);
#pragma unroll
      for (int t = 0; t < 2; ++t) {
        v8f z = {};
        z = wmma16x16x32(qa[t][0], k0, z);
        z = wmma16x16x32(qa[t][1], k1, z);
        sc[t][u] = z;
      }
    }
#pragma unroll
    for (int t = 0; t < 2; ++t)
#pragma unroll
      for (int u = 0; u < 2; ++u)
#pragma unroll
        for (int r = 0; r < 8; ++r) {
          const int q = q0 + 16 * t + r + 8 * hf;
          const int k = kb + 16 * u + nl;
          const float s = sc[t][u][r] * scale;
          sc[t][u][r] = (k <= q) ? s : -1e30f;
        }
#pragma unroll
    for (int t = 0; t < 2; ++t) {
      float alpha[8];
#pragma unroll
      for (int r = 0; r < 8; ++r) {
        float v = fmaxf(sc[t][0][r], sc[t][1][r]);
#pragma unroll
        for (int m = 1; m < 16; m <<= 1) v = fmaxf(v, __shfl_xor(v, m, 32));
        const float mnew = fmaxf(rowm[t][r], v);
        alpha[r] = __expf(rowm[t][r] - mnew);
        rowm[t][r] = mnew;
        const float p0 = __expf(sc[t][0][r] - mnew);
        const float p1 = __expf(sc[t][1][r] - mnew);
        sc[t][0][r] = p0;
        sc[t][1][r] = p1;
        float rs = p0 + p1;
#pragma unroll
        for (int m = 1; m < 16; m <<= 1) rs += __shfl_xor(rs, m, 32);
        rowl[t][r] = rowl[t][r] * alpha[r] + rs;
      }
#pragma unroll
      for (int j = 0; j < 4; ++j)
#pragma unroll
        for (int r = 0; r < 8; ++r) o[t][j][r] *= alpha[r];
#pragma unroll
      for (int u = 0; u < 2; ++u)
#pragma unroll
        for (int r = 0; r < 8; ++r)
          pbuf[wave][(16 * t + r + 8 * hf) * 32 + 16 * u + nl] =
              (_Float16)sc[t][u][r];
    }
    v16h pa[2];
#pragma unroll
    for (int t = 0; t < 2; ++t)
      pa[t] = load_tile_f16(&pbuf[wave][(16 * t) * 32], 32, lane);
#pragma unroll
    for (int j = 0; j < 4; ++j) {
      v16h vb = load_tile_f16(Vt + baseV + (size_t)(16 * j) * SK_ + kb, SK_, lane);
#pragma unroll
      for (int t = 0; t < 2; ++t) o[t][j] = wmma16x16x32(pa[t], vb, o[t][j]);
    }
  }
#pragma unroll
  for (int t = 0; t < 2; ++t)
#pragma unroll
    for (int j = 0; j < 4; ++j)
#pragma unroll
      for (int r = 0; r < 8; ++r) {
        const int q = q0 + 16 * t + r + 8 * hf;
        const float val = o[t][j][r] / rowl[t][r];
        Of[(size_t)(bb * SQ_ + q) * H_ + hh * 64 + 16 * j + nl] = (_Float16)val;
      }
}

// ---------------- Stage 4: residual + LayerNorm ----------------------------
__global__ __launch_bounds__(256) void ln_kernel(
    const float* __restrict__ Y, const float* __restrict__ X,
    const float* __restrict__ g, const float* __restrict__ bta,
    float* __restrict__ out) {
  const int row = blockIdx.x;
  const float* y = Y + (size_t)row * H_;
  const float* x = X + (size_t)row * H_;
  __shared__ float red0[8], red1[8];
  float s = 0.0f, s2 = 0.0f, vals[4];
#pragma unroll
  for (int i = 0; i < 4; ++i) {
    const int c = threadIdx.x + 256 * i;
    const float v = x[c] + y[c];
    vals[i] = v;
    s += v;
    s2 += v * v;
  }
#pragma unroll
  for (int m = 1; m < 32; m <<= 1) {
    s += __shfl_xor(s, m, 32);
    s2 += __shfl_xor(s2, m, 32);
  }
  const int wave = threadIdx.x >> 5, lane = threadIdx.x & 31;
  if (lane == 0) { red0[wave] = s; red1[wave] = s2; }
  __syncthreads();
  float ts = 0.0f, ts2 = 0.0f;
#pragma unroll
  for (int i = 0; i < 8; ++i) { ts += red0[i]; ts2 += red1[i]; }
  const float mu = ts * (1.0f / H_);
  const float var = ts2 * (1.0f / H_) - mu * mu;
  const float inv = rsqrtf(var + 1e-5f);
#pragma unroll
  for (int i = 0; i < 4; ++i) {
    const int c = threadIdx.x + 256 * i;
    out[(size_t)row * H_ + c] = (vals[i] - mu) * inv * g[c] + bta[c];
  }
}

extern "C" void kernel_launch(void* const* d_in, const int* in_sizes, int n_in,
                              void* d_out, int out_size, void* d_ws, size_t ws_size,
                              hipStream_t stream) {
  const float* query = (const float*)d_in[0];
  const float* key   = (const float*)d_in[1];
  const float* value = (const float*)d_in[2];
  // d_in[3] = causal_mask, computed analytically (tril) in-kernel
  const float* Wq = (const float*)d_in[4];
  const float* bq = (const float*)d_in[5];
  const float* Wk = (const float*)d_in[6];
  const float* bk = (const float*)d_in[7];
  const float* Wv = (const float*)d_in[8];
  const float* bv = (const float*)d_in[9];
  const float* Wo = (const float*)d_in[10];
  const float* bo = (const float*)d_in[11];
  const float* lg = (const float*)d_in[12];
  const float* lb = (const float*)d_in[13];

  char* ws = (char*)d_ws;
  const size_t MB = 1024 * 1024;
  _Float16* Xq16 = (_Float16*)(ws + 0 * MB);    // 8 MB each
  _Float16* Xk16 = (_Float16*)(ws + 8 * MB);
  _Float16* Xv16 = (_Float16*)(ws + 16 * MB);
  _Float16* Wq16 = (_Float16*)(ws + 24 * MB);   // 2 MB each
  _Float16* Wk16 = (_Float16*)(ws + 26 * MB);
  _Float16* Wv16 = (_Float16*)(ws + 28 * MB);
  _Float16* Wo16 = (_Float16*)(ws + 30 * MB);
  _Float16* Qf   = (_Float16*)(ws + 32 * MB);
  _Float16* Kf   = (_Float16*)(ws + 40 * MB);
  _Float16* Vt   = (_Float16*)(ws + 48 * MB);
  _Float16* Of   = (_Float16*)(ws + 56 * MB);
  float* Ytmp = (float*)(ws);  // overlays Xq16/Xk16 after attention (stream-ordered)

  const int NX8 = (B_ * SQ_ * H_) / 8;  // 524288
  const int NW8 = (H_ * H_) / 8;        // 131072
  cvt_kernel<<<NX8 / 256, 256, 0, stream>>>(query, Xq16, NX8);
  cvt_kernel<<<NX8 / 256, 256, 0, stream>>>(key, Xk16, NX8);
  cvt_kernel<<<NX8 / 256, 256, 0, stream>>>(value, Xv16, NX8);
  cvt_kernel<<<NW8 / 256, 256, 0, stream>>>(Wq, Wq16, NW8);
  cvt_kernel<<<NW8 / 256, 256, 0, stream>>>(Wk, Wk16, NW8);
  cvt_kernel<<<NW8 / 256, 256, 0, stream>>>(Wv, Wv16, NW8);
  cvt_kernel<<<NW8 / 256, 256, 0, stream>>>(Wo, Wo16, NW8);

  dim3 gg(H_ / 64, (B_ * SQ_) / 128);
  gemm_kernel<<<gg, 128, 0, stream>>>(Xq16, Wq16, bq, Qf, nullptr, 0);
  gemm_kernel<<<gg, 128, 0, stream>>>(Xk16, Wk16, bk, Kf, nullptr, 0);
  gemm_kernel<<<gg, 128, 0, stream>>>(Xv16, Wv16, bv, Vt, nullptr, 1);
  attn_kernel<<<dim3(SQ_ / 128, NH_, B_), 128, 0, stream>>>(Qf, Kf, Vt, Of);
  gemm_kernel<<<gg, 128, 0, stream>>>(Of, Wo16, bo, nullptr, Ytmp, 2);
  ln_kernel<<<B_ * SQ_, 256, 0, stream>>>(Ytmp, query, lg, lb, (float*)d_out);
}
